// Concat_Linear_62019327754368
// MI455X (gfx1250) — compile-verified, matrix operand order checked by reading, not run on hardware
//
#include <hip/hip_runtime.h>
#include <stdint.h>

// ---------------------------------------------------------------------------
// Fused kernel for:  last = feat[:,0,:] @ W_int.T ; this = feat[:,1,:] @ W_stim.T
//                    bil  = einsum('ba,bj,ajk->bk', this, last, trans)
//                    out  = concat(this, LN(bil)) @ W_out.T + b_out
//
// MI455X reasoning: feat is 402.7 MB fp32 -> pure HBM-bound, ~17.4 us floor at
// 23.3 TB/s. The two [B,768]x[768,9] GEMMs need ~106 TFLOP-equivalent over that
// window -> run them on v_wmma_f32_16x16x32_bf16 (N padded 9->16). Weights are
// zero-padded to 16 rows in LDS so the B-fragment path is branch-free, and
// fp32->bf16 conversion is 2x v_add + v_perm_b32 per packed pair (1.5 VALU
// ops/elem). Everything is fused so feat is read exactly once.
// ---------------------------------------------------------------------------

typedef __attribute__((ext_vector_type(16))) __bf16          v16bf;
typedef __attribute__((ext_vector_type(8)))  float           v8f;

#define DIM              768
#define NOUT             9
#define NPAD             16                                  // WMMA N tile
#define SAMPLES_PER_WAVE 16
#define WAVES_PER_BLOCK  8
#define SAMPLES_PER_BLK  (SAMPLES_PER_WAVE * WAVES_PER_BLOCK)   // 128
#define SSTRIDE          10   // result staging stride (floats); conflict-free

__device__ __forceinline__ unsigned short f32_to_bf16_rne(float f) {
    union { float f; uint32_t u; } in; in.f = f;
    uint32_t u = in.u;
    u += 0x7FFFu + ((u >> 16) & 1u);   // round-to-nearest-even (staging path)
    return (unsigned short)(u >> 16);
}

// Pack two floats into {bf16(hi),bf16(lo)} with round-to-nearest (ties away):
// one v_add per float + one v_perm_b32 selecting the two high halves.
__device__ __forceinline__ uint32_t pack2_bf16(float lo, float hi) {
    union { float f; uint32_t u; } a, b; a.f = lo; b.f = hi;
    uint32_t ua = a.u + 0x8000u;
    uint32_t ub = b.u + 0x8000u;
    // perm sel 0x07060302: result = { ub[31:16], ua[31:16] }
    return __builtin_amdgcn_perm(ub, ua, 0x07060302u);
}

struct Frag {
    union { uint32_t d[8]; uint4 q[2]; v16bf v; };
};

__device__ __forceinline__ v8f wmma_bf16(v16bf a, v16bf b, v8f c) {
    // (neg_a, A, neg_b, B, c_mod, C, reuse_a, reuse_b)
    return __builtin_amdgcn_wmma_f32_16x16x32_bf16(false, a, false, b,
                                                   (short)0, c, false, false);
}

// A-fragment: 16-bit A layout (ISA 7.12.2). Lane<16 holds row M=lane,
// Ks kb+0..7 (elems 0..7) and kb+16..23 (elems 8..15); lane>=16 same row,
// kb shifted by +8 (caller passes kb = kk + (lane>=16 ? 8 : 0)).
__device__ __forceinline__ Frag load_a_frag(const float* __restrict__ row, int kb) {
    float4 a0 = *reinterpret_cast<const float4*>(row + kb);
    float4 a1 = *reinterpret_cast<const float4*>(row + kb + 4);
    float4 a2 = *reinterpret_cast<const float4*>(row + kb + 16);
    float4 a3 = *reinterpret_cast<const float4*>(row + kb + 20);
    Frag f;
    f.d[0] = pack2_bf16(a0.x, a0.y);
    f.d[1] = pack2_bf16(a0.z, a0.w);
    f.d[2] = pack2_bf16(a1.x, a1.y);
    f.d[3] = pack2_bf16(a1.z, a1.w);
    f.d[4] = pack2_bf16(a2.x, a2.y);
    f.d[5] = pack2_bf16(a2.z, a2.w);
    f.d[6] = pack2_bf16(a3.x, a3.y);
    f.d[7] = pack2_bf16(a3.z, a3.w);
    return f;
}

// B-fragment: B = W^T (K x 16); lane's column n = lane&15 is weight row n.
// Weight rows 9..15 are zero-padded in LDS -> branch-free ds_load_b128 pair.
__device__ __forceinline__ Frag load_b_frag(const unsigned short* __restrict__ wlds,
                                            int n, int kb) {
    const unsigned short* wr = wlds + (size_t)n * DIM;
    Frag f;
    f.q[0] = *reinterpret_cast<const uint4*>(wr + kb);
    f.q[1] = *reinterpret_cast<const uint4*>(wr + kb + 16);
    return f;
}

__global__ __launch_bounds__(256)
void fused_bilinear_ln_out(const float* __restrict__ feat,
                           const float* __restrict__ W_int,
                           const float* __restrict__ W_stim,
                           const float* __restrict__ trans,
                           const float* __restrict__ ln_w,
                           const float* __restrict__ ln_b,
                           const float* __restrict__ W_out,
                           const float* __restrict__ b_out,
                           float* __restrict__ out,
                           int B)
{
    __shared__ __align__(16) unsigned short sWbf[2][NPAD][DIM]; // bf16, rows 9..15 = 0
    __shared__ float sTrans[9 * 9 * 9];
    __shared__ float sWout[9][18];
    __shared__ float sBout[9], sLnW[9], sLnB[9];
    __shared__ float sLast[SAMPLES_PER_BLK][SSTRIDE];
    __shared__ float sThis[SAMPLES_PER_BLK][SSTRIDE];

    const int tid = threadIdx.x;

    // ---- stage small tensors into LDS (once per block) ----
    unsigned short* wflat = &sWbf[0][0][0];
    for (int i = tid; i < 2 * NPAD * DIM; i += 256) {
        const int w   = i / (NPAD * DIM);
        const int rem = i - w * (NPAD * DIM);
        const int r   = rem / DIM;
        const int k   = rem - r * DIM;
        float v = 0.0f;
        if (r < NOUT) v = w ? W_stim[r * DIM + k] : W_int[r * DIM + k];
        wflat[i] = f32_to_bf16_rne(v);
    }
    for (int i = tid; i < 9 * 9 * 9; i += 256) sTrans[i] = trans[i];
    for (int i = tid; i < 9 * 18; i += 256)    (&sWout[0][0])[i] = W_out[i];
    if (tid < 9) { sBout[tid] = b_out[tid]; sLnW[tid] = ln_w[tid]; sLnB[tid] = ln_b[tid]; }
    __syncthreads();

    // ---- WMMA GEMMs: one 16-sample tile per wave, K=768 in chunks of 32 ----
    const int wave = tid >> 5;
    const int lane = tid & 31;
    const int half = lane >> 4;      // 0: K sub-range +0, 1: +8
    const int l16  = lane & 15;      // A: row M; B: column N
    const int kLaneOff = half * 8;

    const size_t blockSample = (size_t)blockIdx.x * SAMPLES_PER_BLK;
    size_t sample = blockSample + (size_t)wave * SAMPLES_PER_WAVE + l16;
    if (sample >= (size_t)B) sample = (size_t)B - 1;   // clamp loads; stores guarded

    const float* row0 = feat + (sample * 2 + 0) * (size_t)DIM;  // plane 0 -> last
    const float* row1 = row0 + DIM;                             // plane 1 -> this

    v8f accL = {};   // last = feat0 @ W_int.T
    v8f accS = {};   // this = feat1 @ W_stim.T

    #pragma unroll 2
    for (int kk = 0; kk < DIM; kk += 32) {
        const int kb = kk + kLaneOff;
        Frag a0 = load_a_frag(row0, kb);
        Frag b0 = load_b_frag(&sWbf[0][0][0], l16, kb);
        accL = wmma_bf16(a0.v, b0.v, accL);
        Frag a1 = load_a_frag(row1, kb);
        Frag b1 = load_b_frag(&sWbf[1][0][0], l16, kb);
        accS = wmma_bf16(a1.v, b1.v, accS);
    }

    // C layout (ISA 7.12.2): lane<16 -> N=lane, VGPR r -> M=r; lane>=16 -> M=8+r.
    if (l16 < NOUT) {
        const int sbase = wave * SAMPLES_PER_WAVE + half * 8;
        #pragma unroll
        for (int r = 0; r < 8; ++r) {
            sLast[sbase + r][l16] = accL[r];
            sThis[sbase + r][l16] = accS[r];
        }
    }
    __syncthreads();

    // ---- per-sample epilogue: trilinear + LayerNorm + output GEMM ----
    if (tid < SAMPLES_PER_BLK) {
        const size_t g = blockSample + tid;
        if (g < (size_t)B) {
            float th[9], la[9];
            #pragma unroll
            for (int i = 0; i < 9; ++i) { th[i] = sThis[tid][i]; la[i] = sLast[tid][i]; }

            float bil[9] = {0.f,0.f,0.f,0.f,0.f,0.f,0.f,0.f,0.f};
            #pragma unroll
            for (int a = 0; a < 9; ++a) {
                const float ca = th[a];
                #pragma unroll
                for (int j = 0; j < 9; ++j) {
                    const float c = ca * la[j];
                    const float* tr = &sTrans[(a * 9 + j) * 9];  // LDS broadcast
                    #pragma unroll
                    for (int k = 0; k < 9; ++k) bil[k] = fmaf(c, tr[k], bil[k]);
                }
            }

            float mu = 0.f;
            #pragma unroll
            for (int i = 0; i < 9; ++i) mu += bil[i];
            mu *= (1.0f / 9.0f);
            float var = 0.f;
            #pragma unroll
            for (int i = 0; i < 9; ++i) { float d = bil[i] - mu; var = fmaf(d, d, var); }
            var *= (1.0f / 9.0f);
            const float inv = rsqrtf(var + 1e-5f);

            float ln[9];
            #pragma unroll
            for (int i = 0; i < 9; ++i) ln[i] = fmaf((bil[i] - mu) * inv, sLnW[i], sLnB[i]);

            float* op = out + g * NOUT;
            #pragma unroll
            for (int k = 0; k < 9; ++k) {
                float acc = sBout[k];
                #pragma unroll
                for (int i = 0; i < 9; ++i) acc = fmaf(th[i], sWout[k][i], acc);
                #pragma unroll
                for (int i = 0; i < 9; ++i) acc = fmaf(ln[i], sWout[k][9 + i], acc);
                op[k] = acc;
            }
        }
    }
}

extern "C" void kernel_launch(void* const* d_in, const int* in_sizes, int n_in,
                              void* d_out, int out_size, void* d_ws, size_t ws_size,
                              hipStream_t stream) {
    const float* feat   = (const float*)d_in[0];
    const float* W_int  = (const float*)d_in[1];
    const float* W_stim = (const float*)d_in[2];
    const float* trans  = (const float*)d_in[3];
    const float* ln_w   = (const float*)d_in[4];
    const float* ln_b   = (const float*)d_in[5];
    const float* W_out  = (const float*)d_in[6];
    const float* b_out  = (const float*)d_in[7];
    float* out = (float*)d_out;

    const int B = in_sizes[0] / (2 * DIM);          // 65536
    const int blocks = (B + SAMPLES_PER_BLK - 1) / SAMPLES_PER_BLK;   // 512

    fused_bilinear_ln_out<<<blocks, 256, 0, stream>>>(
        feat, W_int, W_stim, trans, ln_w, ln_b, W_out, b_out, out, B);
}
